// LinearAttention_56813827392082
// MI455X (gfx1250) — compile-verified, matrix operand order checked by reading, not run on hardware
//
#include <hip/hip_runtime.h>
#include <hip/hip_bf16.h>
#include <math.h>

// ---------------------------------------------------------------------------
// LinearAttention for MI455X (gfx1250), bf16 WMMA pipeline.
//   B=4, T=8192, C=1024, H=16, d=64
// Round 3: block-local weight-tile sharing (rt fastest in wave mapping),
// k-loop unroll 2 in the big GEMMs, full-lane D_inv reduction.
// ---------------------------------------------------------------------------

typedef __bf16 bf16_t;
typedef bf16_t v16bf __attribute__((ext_vector_type(16)));
typedef bf16_t v8bf  __attribute__((ext_vector_type(8)));
typedef float  v8f   __attribute__((ext_vector_type(8)));

#define B_   4
#define T_   8192
#define C_   1024
#define H_   16
#define D_   64
#define BT_  (B_ * T_)
#define CSPLIT 32   // split-K factor for the context GEMM

// ---- WMMA fragment loaders -------------------------------------------------
// 16-bit A/B layout (CDNA5 ISA): lane l holds row (l&15); lanes
// 0-15 carry K = {k0..k0+7, k0+16..k0+23}, lanes 16-31 carry
// K = {k0+8..k0+15, k0+24..k0+31}.  So each fragment is two contiguous
// 16-byte chunks at (k0 + hi*8) and (k0 + 16 + hi*8).

static __device__ __forceinline__ v16bf frag_bf(const bf16_t* __restrict__ base,
                                                long ld, long row, int k0, int hi) {
  const bf16_t* p = base + row * ld + k0 + (hi << 3);
  v8bf lo = *(const v8bf*)p;
  v8bf hh = *(const v8bf*)(p + 16);
  v16bf r;
#pragma unroll
  for (int i = 0; i < 8; ++i) { r[i] = lo[i]; r[i + 8] = hh[i]; }
  return r;
}

static __device__ __forceinline__ v16bf frag_f32(const float* __restrict__ base,
                                                 long ld, long row, int k0, int hi) {
  const float* p = base + row * ld + k0 + (hi << 3);
  v16bf r;
#pragma unroll
  for (int i = 0; i < 8; ++i) { r[i] = (bf16_t)p[i]; r[i + 8] = (bf16_t)p[16 + i]; }
  return r;
}

static __device__ __forceinline__ v8f wmma_bf(v16bf a, v16bf b, v8f c) {
  return __builtin_amdgcn_wmma_f32_16x16x32_bf16(false, a, false, b, (short)0, c,
                                                 false, false);
}

// ---- small helper kernels --------------------------------------------------

__global__ void cvt_f32_bf16(const float* __restrict__ src, bf16_t* __restrict__ dst,
                             int n) {
  int i = blockIdx.x * blockDim.x + threadIdx.x;
  if (i < n) dst[i] = (bf16_t)src[i];
}

// vectorized variant for the big x conversion (4 elems/thread)
__global__ void cvt4_f32_bf16(const float* __restrict__ src, bf16_t* __restrict__ dst,
                              int n4) {
  int i = blockIdx.x * blockDim.x + threadIdx.x;
  if (i < n4) {
    float4 f = ((const float4*)src)[i];
    typedef bf16_t v4bf __attribute__((ext_vector_type(4)));
    v4bf o;
    o[0] = (bf16_t)f.x; o[1] = (bf16_t)f.y; o[2] = (bf16_t)f.z; o[3] = (bf16_t)f.w;
    ((v4bf*)dst)[i] = o;
  }
}

__global__ void zero_f32(float* __restrict__ p, int n) {
  int i = blockIdx.x * blockDim.x + threadIdx.x;
  if (i < n) p[i] = 0.0f;
}

// ---- QKV projection + feature-map softmax ----------------------------------
// One wave = 32 tokens x 64 cols (one head) for one of {q,k,v}.
// rt is the fastest wave index: the 8 waves of a block share (w,h), so the
// 64xK weight tile is fetched once per WGP and reused from cache.
// K-loop: 32 steps, each = 2 A-frags + 4 B-frags -> 8 v_wmma_f32_16x16x32_bf16.
// D-tile layout: lane l, vgpr j -> (M = j + 8*(l>>4), N = l&15).

__global__ __launch_bounds__(256) void qkv_kernel(
    const bf16_t* __restrict__ xbf,
    const bf16_t* __restrict__ Wqb, const bf16_t* __restrict__ Wkb,
    const bf16_t* __restrict__ Wvb,
    const float* __restrict__ bq, const float* __restrict__ bk,
    const float* __restrict__ bv,
    bf16_t* __restrict__ Qs,   // [BT, C] row-major, head-interleaved (bf16)
    bf16_t* __restrict__ KsT,  // [B*H, d, T] (bf16)
    bf16_t* __restrict__ VT,   // [B*H, d, T] (bf16)
    float* __restrict__ ksum)  // [B*H, d] (f32, atomics)
{
  const int lane = threadIdx.x & 31;
  const int nl = lane & 15;
  const int hi = lane >> 4;
  const long gw = (long)blockIdx.x * (blockDim.x >> 5) + (threadIdx.x >> 5);
  const long total = 3l * H_ * (BT_ / 32);
  if (gw >= total) return;

  const long rt = gw % (BT_ / 32);                 // fastest: token tile
  const int  h  = (int)((gw / (BT_ / 32)) % H_);   // shared within a block
  const int  w  = (int)(gw / ((long)H_ * (BT_ / 32)));

  const bf16_t* Wb = (w == 0) ? Wqb : (w == 1) ? Wkb : Wvb;
  const float*  bb = (w == 0) ? bq  : (w == 1) ? bk  : bv;

  const long row0 = rt * 32;       // 32 tokens per wave (2 M-tiles)
  const int  n0   = h * D_;

  v8f acc[2][4];
#pragma unroll
  for (int mt = 0; mt < 2; ++mt)
#pragma unroll
    for (int nt = 0; nt < 4; ++nt)
#pragma unroll
      for (int j = 0; j < 8; ++j) acc[mt][nt][j] = 0.0f;

#pragma unroll 2
  for (int k0 = 0; k0 < C_; k0 += 32) {
    v16bf a0 = frag_bf(xbf, C_, row0 + nl, k0, hi);
    v16bf a1 = frag_bf(xbf, C_, row0 + 16 + nl, k0, hi);
#pragma unroll
    for (int nt = 0; nt < 4; ++nt) {
      v16bf b = frag_bf(Wb, C_, n0 + nt * 16 + nl, k0, hi);  // B^T rows = W rows
      acc[0][nt] = wmma_bf(a0, b, acc[0][nt]);
      acc[1][nt] = wmma_bf(a1, b, acc[1][nt]);
    }
  }

  // bias
#pragma unroll
  for (int nt = 0; nt < 4; ++nt) {
    float bval = bb[n0 + nt * 16 + nl];
#pragma unroll
    for (int mt = 0; mt < 2; ++mt)
#pragma unroll
      for (int j = 0; j < 8; ++j) acc[mt][nt][j] += bval;
  }

  // feature-map softmax over the 64-wide head dim (rows live across 16 lanes
  // of each half-wave; width-16 xor reductions keep halves independent)
  if (w < 2) {
#pragma unroll
    for (int mt = 0; mt < 2; ++mt)
#pragma unroll
      for (int j = 0; j < 8; ++j) {
        float m = fmaxf(fmaxf(acc[mt][0][j], acc[mt][1][j]),
                        fmaxf(acc[mt][2][j], acc[mt][3][j]));
        m = fmaxf(m, __shfl_xor(m, 1, 16));
        m = fmaxf(m, __shfl_xor(m, 2, 16));
        m = fmaxf(m, __shfl_xor(m, 4, 16));
        m = fmaxf(m, __shfl_xor(m, 8, 16));
        float s = 0.0f;
#pragma unroll
        for (int nt = 0; nt < 4; ++nt) {
          float e = __expf(acc[mt][nt][j] - m);
          acc[mt][nt][j] = e;
          s += e;
        }
        s += __shfl_xor(s, 1, 16);
        s += __shfl_xor(s, 2, 16);
        s += __shfl_xor(s, 4, 16);
        s += __shfl_xor(s, 8, 16);
        float r = 1.0f / s;
#pragma unroll
        for (int nt = 0; nt < 4; ++nt) acc[mt][nt][j] *= r;
      }
  }

  const long bidx = row0 / T_;
  const long t0   = row0 % T_;
  const long bh   = bidx * H_ + h;

  if (w == 0) {
#pragma unroll
    for (int mt = 0; mt < 2; ++mt)
#pragma unroll
      for (int nt = 0; nt < 4; ++nt)
#pragma unroll
        for (int j = 0; j < 8; ++j) {
          long grow = row0 + mt * 16 + j + 8 * hi;
          Qs[grow * C_ + n0 + nt * 16 + nl] = (bf16_t)acc[mt][nt][j];
        }
  } else {
    bf16_t* dstT = (w == 1) ? KsT : VT;
#pragma unroll
    for (int nt = 0; nt < 4; ++nt) {
      float part = 0.0f;
#pragma unroll
      for (int mt = 0; mt < 2; ++mt)
#pragma unroll
        for (int j = 0; j < 8; ++j) {
          long t = t0 + mt * 16 + j + 8 * hi;
          dstT[(bh * D_ + nt * 16 + nl) * (long)T_ + t] = (bf16_t)acc[mt][nt][j];
          part += acc[mt][nt][j];
        }
      if (w == 1) atomicAdd(&ksum[bh * D_ + nt * 16 + nl], part);
    }
  }
}

// ---- context = k^T v per (b,h), split-K over T, transposed f32 output ------

__global__ __launch_bounds__(256) void ctx_kernel(
    const bf16_t* __restrict__ KsT, const bf16_t* __restrict__ VT,
    float* __restrict__ ctxT)  // [B*H, 64(N), 64(M)]  (= context transposed)
{
  const int lane = threadIdx.x & 31;
  const int nl = lane & 15;
  const int hi = lane >> 4;
  const long gw = (long)blockIdx.x * (blockDim.x >> 5) + (threadIdx.x >> 5);
  const long total = (long)CSPLIT * (B_ * H_) * 4;
  if (gw >= total) return;

  const int mt = (int)(gw % 4);
  const int bh = (int)((gw / 4) % (B_ * H_));
  const int sp = (int)(gw / (4 * B_ * H_));
  const int chunk = T_ / CSPLIT;  // 256
  const int kbeg = sp * chunk;

  const long arow = (long)bh * D_ + mt * 16 + nl;

  v8f acc[4];
#pragma unroll
  for (int nt = 0; nt < 4; ++nt)
#pragma unroll
    for (int j = 0; j < 8; ++j) acc[nt][j] = 0.0f;

#pragma unroll 2
  for (int k0 = kbeg; k0 < kbeg + chunk; k0 += 32) {
    v16bf a = frag_bf(KsT, T_, arow, k0, hi);
#pragma unroll
    for (int nt = 0; nt < 4; ++nt) {
      v16bf b = frag_bf(VT, T_, (long)bh * D_ + nt * 16 + nl, k0, hi);
      acc[nt] = wmma_bf(a, b, acc[nt]);
    }
  }

#pragma unroll
  for (int nt = 0; nt < 4; ++nt)
#pragma unroll
    for (int j = 0; j < 8; ++j) {
      int M = mt * 16 + j + 8 * hi;       // context row (d index)
      int N = nt * 16 + nl;               // context col (v index)
      atomicAdd(&ctxT[(long)bh * D_ * D_ + (long)N * D_ + M], acc[nt][j]);
    }
}

// ---- out = (q @ context) * D_inv + q  (per head, bf16 output) --------------

__global__ __launch_bounds__(256) void attn_out_kernel(
    const bf16_t* __restrict__ Qs, const float* __restrict__ ctxT,
    const float* __restrict__ ksum, bf16_t* __restrict__ ybf)
{
  const int lane = threadIdx.x & 31;
  const int nl = lane & 15;
  const int hi = lane >> 4;
  const long gw = (long)blockIdx.x * (blockDim.x >> 5) + (threadIdx.x >> 5);
  const long total = (long)(B_ * H_) * (T_ / 16);
  if (gw >= total) return;

  const int  tt = (int)(gw % (T_ / 16));     // fastest: block shares bh
  const int  bh = (int)(gw / (T_ / 16));
  const int  b  = bh / H_;
  const int  h  = bh % H_;
  const long row0 = (long)b * T_ + (long)tt * 16;
  const int  n0h  = h * D_;

  v8f acc[4];
#pragma unroll
  for (int nt = 0; nt < 4; ++nt)
#pragma unroll
    for (int j = 0; j < 8; ++j) acc[nt][j] = 0.0f;

#pragma unroll
  for (int k0 = 0; k0 < D_; k0 += 32) {
    v16bf a = frag_bf(Qs + n0h, C_, row0 + nl, k0, hi);
#pragma unroll
    for (int nt = 0; nt < 4; ++nt) {
      v16bf bfr = frag_f32(ctxT + (long)bh * D_ * D_, D_, nt * 16 + nl, k0, hi);
      acc[nt] = wmma_bf(a, bfr, acc[nt]);
    }
  }

  // D_inv for row (lane&15): lane r covers d in [0,32), lane r+16 covers
  // [32,64); combine halves with one xor-16 shuffle -> all lanes hold the
  // full dot for their row.
  float s = 0.0f;
  {
    const bf16_t* qp = Qs + (row0 + nl) * C_ + n0h + hi * 32;
    const float*  kp = ksum + (long)bh * D_ + hi * 32;
#pragma unroll 8
    for (int d = 0; d < 32; ++d) s += (float)qp[d] * kp[d];
  }
  s += __shfl_xor(s, 16, 32);
  float dinv = 1.0f / s;

#pragma unroll
  for (int j = 0; j < 8; ++j) {
    float dv = __shfl(dinv, j + 8 * hi, 32);   // row j+8*hi
    long grow = row0 + j + 8 * hi;
#pragma unroll
    for (int nt = 0; nt < 4; ++nt) {
      int col = n0h + nt * 16 + nl;
      float qv = (float)Qs[grow * C_ + col];
      ybf[grow * C_ + col] = (bf16_t)(acc[nt][j] * dv + qv);
    }
  }
}

// ---- final projection: out = y @ Wp^T + bp (fp32 output), M=32 -------------

__global__ __launch_bounds__(256) void proj_kernel(
    const bf16_t* __restrict__ ybf, const bf16_t* __restrict__ Wpb,
    const float* __restrict__ bp, float* __restrict__ out)
{
  const int lane = threadIdx.x & 31;
  const int nl = lane & 15;
  const int hi = lane >> 4;
  const long gw = (long)blockIdx.x * (blockDim.x >> 5) + (threadIdx.x >> 5);
  const long total = (long)(BT_ / 32) * (C_ / 64);
  if (gw >= total) return;

  const long rt = gw % (BT_ / 32);             // fastest: block shares nc
  const int  nc = (int)(gw / (BT_ / 32));
  const long row0 = rt * 32;
  const int  n0 = nc * 64;

  v8f acc[2][4];
#pragma unroll
  for (int mt = 0; mt < 2; ++mt)
#pragma unroll
    for (int nt = 0; nt < 4; ++nt)
#pragma unroll
      for (int j = 0; j < 8; ++j) acc[mt][nt][j] = 0.0f;

#pragma unroll 2
  for (int k0 = 0; k0 < C_; k0 += 32) {
    v16bf a0 = frag_bf(ybf, C_, row0 + nl, k0, hi);
    v16bf a1 = frag_bf(ybf, C_, row0 + 16 + nl, k0, hi);
#pragma unroll
    for (int nt = 0; nt < 4; ++nt) {
      v16bf b = frag_bf(Wpb, C_, n0 + nt * 16 + nl, k0, hi);
      acc[0][nt] = wmma_bf(a0, b, acc[0][nt]);
      acc[1][nt] = wmma_bf(a1, b, acc[1][nt]);
    }
  }

#pragma unroll
  for (int nt = 0; nt < 4; ++nt) {
    float bval = bp[n0 + nt * 16 + nl];
#pragma unroll
    for (int mt = 0; mt < 2; ++mt)
#pragma unroll
      for (int j = 0; j < 8; ++j) {
        long grow = row0 + mt * 16 + j + 8 * hi;
        out[grow * C_ + n0 + nt * 16 + nl] = acc[mt][nt][j] + bval;
      }
  }
}

// ---------------------------------------------------------------------------

extern "C" void kernel_launch(void* const* d_in, const int* in_sizes, int n_in,
                              void* d_out, int out_size, void* d_ws, size_t ws_size,
                              hipStream_t stream) {
  const float* x  = (const float*)d_in[0];
  const float* Wq = (const float*)d_in[1];
  const float* bq = (const float*)d_in[2];
  const float* Wk = (const float*)d_in[3];
  const float* bk = (const float*)d_in[4];
  const float* Wv = (const float*)d_in[5];
  const float* bv = (const float*)d_in[6];
  const float* Wp = (const float*)d_in[7];
  const float* bp = (const float*)d_in[8];
  float* out = (float*)d_out;
  char* ws = (char*)d_ws;

  const size_t MB = 1024ull * 1024ull;
  bf16_t* WqB  = (bf16_t*)(ws + 0 * MB);
  bf16_t* WkB  = (bf16_t*)(ws + 2 * MB);
  bf16_t* WvB  = (bf16_t*)(ws + 4 * MB);
  bf16_t* WpB  = (bf16_t*)(ws + 6 * MB);
  float*  ksum = (float*)(ws + 8 * MB);              // 4096 f32
  float*  ctxT = (float*)(ws + 8 * MB + 16384);      // 262144 f32
  bf16_t* Qs   = (bf16_t*)(ws + 16 * MB);            // 64 MiB
  bf16_t* KsT  = (bf16_t*)(ws + 80 * MB);            // 64 MiB
  bf16_t* VT   = (bf16_t*)(ws + 144 * MB);           // 64 MiB
  bf16_t* xbf  = (bf16_t*)(ws + 208 * MB);           // 64 MiB (live only in QKV)
  bf16_t* ybf  = KsT;  // alias: KsT dead after ctx_kernel

  // 1) conversions (weights + x) and accumulator zeroing
  const int nW = C_ * C_;
  cvt_f32_bf16<<<(nW + 255) / 256, 256, 0, stream>>>(Wq, WqB, nW);
  cvt_f32_bf16<<<(nW + 255) / 256, 256, 0, stream>>>(Wk, WkB, nW);
  cvt_f32_bf16<<<(nW + 255) / 256, 256, 0, stream>>>(Wv, WvB, nW);
  cvt_f32_bf16<<<(nW + 255) / 256, 256, 0, stream>>>(Wp, WpB, nW);
  {
    int n4 = (BT_ * C_) / 4;  // 8388608
    cvt4_f32_bf16<<<(n4 + 255) / 256, 256, 0, stream>>>(x, xbf, n4);
  }
  zero_f32<<<(4096 + 262144 + 255) / 256, 256, 0, stream>>>(ksum, 4096 + 262144);

  // 2) q/k/v projections (+softmax feature map, k_sum)
  {
    long waves = 3l * H_ * (BT_ / 32);           // 49152
    int blocks = (int)((waves + 7) / 8);         // 6144
    qkv_kernel<<<blocks, 256, 0, stream>>>(xbf, WqB, WkB, WvB, bq, bk, bv,
                                           Qs, KsT, VT, ksum);
  }

  // 3) context = k^T v (split-K atomics)
  {
    long waves = (long)CSPLIT * (B_ * H_) * 4;   // 8192
    int blocks = (int)((waves + 7) / 8);         // 1024
    ctx_kernel<<<blocks, 256, 0, stream>>>(KsT, VT, ctxT);
  }

  // 4) out = q@context * D_inv + q
  {
    long waves = (long)(B_ * H_) * (T_ / 16);    // 32768
    int blocks = (int)((waves + 7) / 8);         // 4096
    attn_out_kernel<<<blocks, 256, 0, stream>>>(Qs, ctxT, ksum, ybf);
  }

  // 5) final projection
  {
    long waves = (long)(BT_ / 32) * (C_ / 64);   // 16384
    int blocks = (int)((waves + 7) / 8);         // 2048
    proj_kernel<<<blocks, 256, 0, stream>>>(ybf, WpB, bp, out);
  }
}
